// AdapterController_55104430408056
// MI455X (gfx1250) — compile-verified, matrix operand order checked by reading, not run on hardware
//
#include <hip/hip_runtime.h>

typedef __attribute__((ext_vector_type(16))) _Float16 v16h;
typedef __attribute__((ext_vector_type(8)))  _Float16 v8h;
typedef __attribute__((ext_vector_type(4)))  _Float16 v4h;
typedef __attribute__((ext_vector_type(8)))  float    v8f;
typedef __attribute__((ext_vector_type(4)))  float    v4f;

// Problem constants (match reference setup_inputs)
constexpr int Mr = 4;      // routers
constexpr int Ne = 16;     // experts per router
constexpr int Cd = 1024;   // model dim
constexpr int Dd = 64;     // down dim
constexpr int Bb = 8;      // batch
constexpr int Ss = 2048;   // seq

constexpr int ROWS = 128;  // S-rows per workgroup
constexpr int KC   = 128;  // C-chunk for GEMM1 staging
constexpr int NC   = 128;  // C-chunk for GEMM2 output

// LDS strides in halves — multiples of 8 so b128 operand loads stay 16B aligned
constexpr int XSTR = 136;  // sA row stride
constexpr int BSTR = 136;  // sB (DW^T) row stride
constexpr int ZSTR = 72;   // sZ row stride
constexpr int USTR = 72;   // sU (UW^T) row stride

constexpr int SA_H = ROWS * XSTR;   // 17408 halves
constexpr int SB_H = Dd   * BSTR;   //  8704 halves
constexpr int SZ_H = ROWS * ZSTR;   //  9216 halves
constexpr int SU_H = NC   * USTR;   //  9216 halves
constexpr int SMEM_H = (SA_H + SB_H) > (SZ_H + SU_H) ? (SA_H + SB_H) : (SZ_H + SU_H);

static __device__ inline v16h ld_op(const _Float16* p0, const _Float16* p1) {
    // Assemble a 16x16x32 operand: 8 halves = K[base+hi*8 .. +8),
    // 8 halves = K[base+16+hi*8 .. +8)  (ISA 16-bit A layout; B mirrored with lane=N)
    v8h lo = *(const v8h*)p0;
    v8h hh = *(const v8h*)p1;
    v16h r;
#pragma unroll
    for (int j = 0; j < 8; ++j) { r[j] = lo[j]; r[j + 8] = hh[j]; }
    return r;
}

static __device__ inline v8f wmma16(v16h a, v16h b, v8f c) {
    return __builtin_amdgcn_wmma_f32_16x16x32_f16(false, a, false, b, (short)0, c, false, false);
}

__global__ __launch_bounds__(256) void adapter_moe_kernel(
    const float* __restrict__ x,      // [B,S,C]
    const float* __restrict__ dw,     // [M,N,C,D]
    const float* __restrict__ db,     // [M,N,D]
    const float* __restrict__ uw,     // [M,N,D,C]
    const int*   __restrict__ eidx,   // [M,B]
    float*       __restrict__ out)    // [M,B,S,C]
{
    __shared__ __align__(16) _Float16 smem[SMEM_H];
    _Float16* sA = smem;              // phase 1: X tile   [ROWS][XSTR]
    _Float16* sB = smem + SA_H;       // phase 1: DW^T     [Dd][BSTR]
    _Float16* sZ = smem;              // phase 2: Z tile   [ROWS][ZSTR]
    _Float16* sU = smem + SZ_H;       // phase 2: UW^T     [NC][USTR]

    const int tid  = threadIdx.x;
    const int lane = tid & 31;
    const int wave = tid >> 5;
    const int lm   = lane & 15;       // A: row-in-stripe, B: column
    const int hi   = lane >> 4;       // lane-half select
    const int wr0  = wave * 16;       // wave's first row in tile

    const int tilesPerSeq = Ss / ROWS;               // 16
    const int t  = blockIdx.x % tilesPerSeq;
    const int mb = blockIdx.x / tilesPerSeq;
    const int m  = mb / Bb;
    const int b  = mb % Bb;
    const int e  = eidx[m * Bb + b];

    const float* xb  = x  + ((size_t)b * Ss + (size_t)t * ROWS) * Cd;
    const float* dwe = dw + (size_t)(m * Ne + e) * Cd * Dd;
    const float* dbe = db + (size_t)(m * Ne + e) * Dd;
    const float* uwe = uw + (size_t)(m * Ne + e) * Dd * Cd;
    float*       ob  = out + ((size_t)(m * Bb + b) * Ss + (size_t)t * ROWS) * Cd;

    v8f acc[4];
#pragma unroll
    for (int i = 0; i < 4; ++i) acc[i] = (v8f){};

    // ---------------- GEMM1: Z[128,64] = X[128,C] * DW_e[C,64] ----------------
    for (int kc = 0; kc < Cd; kc += KC) {
        // Stage X chunk fp32 -> f16 row-major [r][k]
#pragma unroll
        for (int it = 0; it < (ROWS * KC / 4) / 256; ++it) {   // 16 iters
            int g  = it * 256 + tid;
            int r  = g >> 5;                 // KC/4 = 32 float4 per row
            int c4 = g & 31;
            v4f v = *(const v4f*)(xb + (size_t)r * Cd + kc + c4 * 4);
            v4h h;
#pragma unroll
            for (int j = 0; j < 4; ++j) h[j] = (_Float16)v[j];
            *(v4h*)(sA + r * XSTR + c4 * 4) = h;
        }
        // Stage DW chunk transposed: global [c][d] -> LDS [d][c]
#pragma unroll
        for (int it = 0; it < (KC * Dd / 4) / 256; ++it) {     // 8 iters
            int g  = it * 256 + tid;
            int c  = g >> 4;                 // Dd/4 = 16 float4 per c-row
            int d4 = g & 15;
            v4f v = *(const v4f*)(dwe + (size_t)(kc + c) * Dd + d4 * 4);
#pragma unroll
            for (int j = 0; j < 4; ++j) sB[(d4 * 4 + j) * BSTR + c] = (_Float16)v[j];
        }
        if (kc + KC < Cd) {  // hint next X chunk toward L2 (global_prefetch_b8)
            __builtin_prefetch(xb + (size_t)(tid >> 1) * Cd + (kc + KC) + (tid & 1) * 64, 0, 0);
        }
        __syncthreads();

        const _Float16* aRow = sA + (wr0 + lm) * XSTR;
#pragma unroll
        for (int ks = 0; ks < KC / 32; ++ks) {                 // 4 K-steps of 32
            const int kb = ks * 32;
            v16h av = ld_op(aRow + kb + hi * 8, aRow + kb + 16 + hi * 8);
#pragma unroll
            for (int tt = 0; tt < 4; ++tt) {                   // D tiles 0..3
                const _Float16* bRow = sB + (tt * 16 + lm) * BSTR;
                v16h bv = ld_op(bRow + kb + hi * 8, bRow + kb + 16 + hi * 8);
                acc[tt] = wmma16(av, bv, acc[tt]);
            }
        }
        __syncthreads();
    }

    // ---------------- bias + swish (f32), Z -> f16 in LDS ----------------
#pragma unroll
    for (int tt = 0; tt < 4; ++tt) {
        const float bv = dbe[tt * 16 + lm];
#pragma unroll
        for (int i = 0; i < 8; ++i) {
            float z = acc[tt][i] + bv;
            z = z / (1.0f + __expf(-z));     // swish
            acc[tt][i] = z;
        }
    }
    // last loop iteration ended with a barrier: safe to overwrite sA region
#pragma unroll
    for (int tt = 0; tt < 4; ++tt) {
#pragma unroll
        for (int i = 0; i < 8; ++i) {
            sZ[(wr0 + 8 * hi + i) * ZSTR + tt * 16 + lm] = (_Float16)acc[tt][i];
        }
    }
    __syncthreads();

    // ---------------- GEMM2: U[128,C] = Z[128,64] * UW_e[64,C] ----------------
    // A operands (this wave's 16 Z-rows, K=64 -> two 16x16x32 steps) are N-invariant
    const _Float16* zRow = sZ + (wr0 + lm) * ZSTR;
    const v16h a0 = ld_op(zRow + 0  + hi * 8, zRow + 16 + hi * 8);
    const v16h a1 = ld_op(zRow + 32 + hi * 8, zRow + 48 + hi * 8);

    for (int nc = 0; nc < Cd; nc += NC) {
        // Stage UW chunk transposed: global [d][c] -> LDS [c][d]
#pragma unroll
        for (int it = 0; it < (Dd * NC / 4) / 256; ++it) {     // 8 iters
            int g  = it * 256 + tid;
            int d  = g >> 5;                 // NC/4 = 32 float4 per d-row
            int c4 = g & 31;
            v4f v = *(const v4f*)(uwe + (size_t)d * Cd + nc + c4 * 4);
#pragma unroll
            for (int j = 0; j < 4; ++j) sU[(c4 * 4 + j) * USTR + d] = (_Float16)v[j];
        }
        if (nc + NC < Cd) {
            __builtin_prefetch(uwe + (size_t)(tid >> 2) * Cd + (nc + NC) + (tid & 3) * 32, 0, 0);
        }
        __syncthreads();

#pragma unroll
        for (int tt = 0; tt < NC / 16; ++tt) {                 // 8 output n-tiles
            const _Float16* uRow = sU + (tt * 16 + lm) * USTR;
            v16h b0 = ld_op(uRow + 0  + hi * 8, uRow + 16 + hi * 8);
            v16h b1 = ld_op(uRow + 32 + hi * 8, uRow + 48 + hi * 8);
            v8f o = (v8f){};
            o = wmma16(a0, b0, o);
            o = wmma16(a1, b1, o);
            // C/D layout: VGPR i holds row (i + 8*hi), column lm (per 16-lane half)
            float* orow = ob + (size_t)(wr0 + 8 * hi) * Cd + nc + tt * 16 + lm;
#pragma unroll
            for (int i = 0; i < 8; ++i) orow[(size_t)i * Cd] = o[i];
        }
        __syncthreads();
    }
}

extern "C" void kernel_launch(void* const* d_in, const int* in_sizes, int n_in,
                              void* d_out, int out_size, void* d_ws, size_t ws_size,
                              hipStream_t stream) {
    const float* x   = (const float*)d_in[0];
    const float* dw  = (const float*)d_in[1];
    const float* db  = (const float*)d_in[2];
    const float* uw  = (const float*)d_in[3];
    const int*   ei  = (const int*)d_in[4];
    float*       out = (float*)d_out;

    dim3 grid(Mr * Bb * (Ss / ROWS));   // 4*8*16 = 512 workgroups
    dim3 block(256);                     // 8 waves (wave32)
    hipLaunchKernelGGL(adapter_moe_kernel, grid, block, 0, stream,
                       x, dw, db, uw, ei, out);
}
